// Attention_27719718929033
// MI455X (gfx1250) — matrix operand order repaired and checked by reading, not compiled, not run on hardware
//
#include <hip/hip_runtime.h>
#include <hip/hip_bf16.h>

#ifndef __has_builtin
#define __has_builtin(x) 0
#endif

typedef __attribute__((ext_vector_type(16))) _Float16     v16h;
typedef __attribute__((ext_vector_type(8)))  _Float16     v8h;
typedef __attribute__((ext_vector_type(8)))  float        v8f;
typedef __attribute__((ext_vector_type(4)))  unsigned int v4u;
typedef __attribute__((ext_vector_type(8)))  int          v8i;
typedef __attribute__((ext_vector_type(4)))  int          v4i;

#define DEVINL __device__ __forceinline__

static constexpr int C = 256;
static constexpr int N = 4096;   // 64*64 tokens
static constexpr int B = 16;

// ---------------------------------------------------------------------------
// WMMA fragment helpers (gfx1250 wave32, 16x16x32 f16 -> f32)
// C/D fragment (v8f): element v, lane l -> row = v + 8*(l>>4), col = l&15
// A frag: lane holds row (l&15); halves 0..15 = K[k0+16g .. +15]
// B frag: lane holds col (l&15); halves 0..15 = K[k0+16g .. +15]
// Both operands stored row-major along K ("NT" GEMM) -> contiguous b128 pairs.
// ---------------------------------------------------------------------------

DEVINL v16h combine(v8h lo, v8h hi) {
  v16h r;
#pragma unroll
  for (int i = 0; i < 8; ++i) { r[i] = lo[i]; r[i + 8] = hi[i]; }
  return r;
}

DEVINL v16h load_a_frag(const _Float16* __restrict__ m, int ld, int row0, int k0, int lane) {
  const _Float16* p = m + (size_t)(row0 + (lane & 15)) * ld + k0 + (lane >> 4) * 16;
  return combine(*(const v8h*)p, *(const v8h*)(p + 8));
}

DEVINL v16h load_b_frag(const _Float16* __restrict__ m, int ld, int col0, int k0, int lane) {
  const _Float16* p = m + (size_t)(col0 + (lane & 15)) * ld + k0 + (lane >> 4) * 16;
  return combine(*(const v8h*)p, *(const v8h*)(p + 8));
}

// B fragment from an LDS tile stored [16 cols][ld halves along K]
DEVINL v16h load_b_lds(const _Float16* sh, int ld, int k0, int lane) {
  const _Float16* p = sh + (lane & 15) * ld + k0 + (lane >> 4) * 16;
  return combine(*(const v8h*)p, *(const v8h*)(p + 8));
}

DEVINL v8f wmma_f(v16h a, v16h b, v8f c) {
  return __builtin_amdgcn_wmma_f32_16x16x32_f16(false, a, false, b, (short)0, c, false, false);
}

DEVINL v8h cvt8(v8f a) {
  v8h r;
#pragma unroll
  for (int i = 0; i < 8; ++i) r[i] = (_Float16)a[i];
  return r;
}

// ---------------------------------------------------------------------------
// Pack kernels
// ---------------------------------------------------------------------------

// x f32 [B][C][N] -> Xt f16 [B][N][C]  (LDS tiled transpose + downconvert)
__global__ void k_pack_xt(const float* __restrict__ x, _Float16* __restrict__ xt) {
  __shared__ float t[16][17];
  int b = blockIdx.z, c0 = blockIdx.y * 16, n0 = blockIdx.x * 16;
  int tx = threadIdx.x, ty = threadIdx.y;
  const float* xb = x + (size_t)b * C * N;
  t[ty][tx] = xb[(size_t)(c0 + ty) * N + (n0 + tx)];
  __syncthreads();
  xt[(size_t)b * N * C + (size_t)(n0 + ty) * C + (c0 + tx)] = (_Float16)t[tx][ty];
}

__global__ void k_pack_w(const float* __restrict__ w, _Float16* __restrict__ o, int n) {
  int i = blockIdx.x * blockDim.x + threadIdx.x;
  if (i < n) o[i] = (_Float16)w[i];
}

// conv_w f32 [O][I][3][3] -> w9 f16 [kh*3+kw][O][I]  (K = I contiguous)
__global__ void k_pack_convw(const float* __restrict__ w, _Float16* __restrict__ w9) {
  int idx = blockIdx.x * blockDim.x + threadIdx.x;
  if (idx >= 9 * C * C) return;
  int i  = idx & 255;
  int o  = (idx >> 8) & 255;
  int kk = idx >> 16;
  w9[idx] = (_Float16)w[(size_t)(o * C + i) * 9 + kk];
}

// ---------------------------------------------------------------------------
// Stage 1: qkv.  OutT[n,m] = sum_c Xt[n,c]*qkv_w[m,c] + qkv_b[m]
// Weight tile (16 rows x 256 K = 8KB, contiguous) staged to LDS via the
// Tensor Data Mover; all 8 waves read B-fragments from LDS.
// Fragment rows = n, cols = m.  m<256 -> Q[c][n]; <512 -> KT[d][n]; else VT[n][d]
// ---------------------------------------------------------------------------
__global__ void k_qkv(const _Float16* __restrict__ xt, const _Float16* __restrict__ wq,
                      const float* __restrict__ qb,
                      _Float16* __restrict__ Q, _Float16* __restrict__ KT,
                      _Float16* __restrict__ VT) {
  __shared__ _Float16 wsh[16 * C];                       // 8KB weight tile
  int b    = blockIdx.z;
  int m0   = blockIdx.y * 16;                            // 0..752
  int wave = threadIdx.x >> 5, lane = threadIdx.x & 31;
  int n0   = (blockIdx.x * 32 + wave * 4) * 16;          // 4 n-tiles per wave

#if __has_builtin(__builtin_amdgcn_tensor_load_to_lds)
  if (wave == 0) {
    // Tensor DMA descriptor: 1-D tile of 4096 f16 (8KB) from wq + m0*256.
    unsigned long long ga = (unsigned long long)(const void*)(wq + (size_t)m0 * C);
    unsigned int la = (unsigned int)(size_t)(void*)wsh;  // LDS byte offset
    v4u g0;
    g0[0] = 1u;                                          // count=1, user D#
    g0[1] = la;                                          // lds_addr
    g0[2] = (unsigned int)(ga & 0xFFFFFFFFu);            // global_addr[31:0]
    g0[3] = (unsigned int)((ga >> 32) & 0x01FFFFFFu) | 0x80000000u;  // addr[56:32] | type=2
    v8i g1;
    g1[0] = 1 << 16;                                     // data_size = 2 bytes
    g1[1] = (int)((196608u & 0xFFFFu) << 16);            // tensor_dim0 lo16 (=0)
    g1[2] = (int)(196608u >> 16) | (1 << 16);            // tensor_dim0 hi | tensor_dim1=1 lo
    g1[3] = (int)(4096u << 16);                          // tile_dim0 = 4096 elements
    g1[4] = 0;                                           // tile_dim1=0 (unused), tile_dim2=0
    g1[5] = (int)196608u;                                // tensor_dim0_stride lo32
    g1[6] = 0;
    g1[7] = 0;
    v4i gz4 = {0, 0, 0, 0};
    v8i gz8 = {0, 0, 0, 0, 0, 0, 0, 0};
    __builtin_amdgcn_tensor_load_to_lds(g0, g1, gz4, gz4, gz8, 0);
    __builtin_amdgcn_s_wait_tensorcnt(0);
  }
  __syncthreads();
#else
  // Fallback: cooperative VMEM -> LDS copy (16 halves per thread)
  {
    int t = threadIdx.x;
    const v8h* src = (const v8h*)(wq + (size_t)m0 * C);
    v8h* dst = (v8h*)wsh;
    dst[t * 2]     = src[t * 2];
    dst[t * 2 + 1] = src[t * 2 + 1];
  }
  __syncthreads();
#endif

  const _Float16* xb = xt + (size_t)b * N * C;
  v8f acc[4] = {{}, {}, {}, {}};
  for (int k0 = 0; k0 < C; k0 += 32) {
    v16h bf = load_b_lds(wsh, C, k0, lane);              // weights from LDS
#pragma unroll
    for (int t = 0; t < 4; ++t) {
      v16h af = load_a_frag(xb, C, n0 + 16 * t, k0, lane);
      acc[t] = wmma_f(af, bf, acc[t]);
    }
  }
  float bias = qb[m0 + (lane & 15)];
#pragma unroll
  for (int t = 0; t < 4; ++t)
#pragma unroll
    for (int v = 0; v < 8; ++v) acc[t][v] += bias;

  int g = lane >> 4;
  int mcol = m0 + (lane & 15);
  if (m0 < 512) {
    _Float16* dst = (m0 < 256)
        ? (Q  + (size_t)b * C * N + (size_t)mcol * N)
        : (KT + (size_t)b * C * N + (size_t)(mcol - 256) * N);
#pragma unroll
    for (int t = 0; t < 4; ++t)
      *(v8h*)(dst + n0 + 16 * t + 8 * g) = cvt8(acc[t]);
  } else {
    _Float16* vb = VT + (size_t)b * N * C;
    int d = mcol - 512;
#pragma unroll
    for (int t = 0; t < 4; ++t)
#pragma unroll
      for (int v = 0; v < 8; ++v)
        vb[(size_t)(n0 + 16 * t + 8 * g + v) * C + d] = (_Float16)acc[t][v];
  }
}

// ---------------------------------------------------------------------------
// Stage 2: logits[c,d] = (1/16) * sum_n Q[c,n]*KT[d,n]   (K = 4096)
// 2 c-tiles x 2 d-tiles per wave -> 1 b128-pair per WMMA.
// ---------------------------------------------------------------------------
__global__ void k_attn(const _Float16* __restrict__ Q, const _Float16* __restrict__ KT,
                       float* __restrict__ lg) {
  int b    = blockIdx.y;
  int c0   = blockIdx.x * 32;
  int wave = threadIdx.x >> 5, lane = threadIdx.x & 31;
  int d0   = wave * 32;
  const _Float16* qm = Q  + (size_t)b * C * N;
  const _Float16* km = KT + (size_t)b * C * N;

  v8f acc[4] = {{}, {}, {}, {}};                         // [ci][di]
  for (int k0 = 0; k0 < N; k0 += 32) {
    v16h a0 = load_a_frag(qm, N, c0,      k0, lane);
    v16h a1 = load_a_frag(qm, N, c0 + 16, k0, lane);
    v16h b0 = load_b_frag(km, N, d0,      k0, lane);
    v16h b1 = load_b_frag(km, N, d0 + 16, k0, lane);
    acc[0] = wmma_f(a0, b0, acc[0]);
    acc[1] = wmma_f(a0, b1, acc[1]);
    acc[2] = wmma_f(a1, b0, acc[2]);
    acc[3] = wmma_f(a1, b1, acc[3]);
  }
  float* lb = lg + (size_t)b * C * C;
  int g = lane >> 4;
#pragma unroll
  for (int ci = 0; ci < 2; ++ci)
#pragma unroll
    for (int di = 0; di < 2; ++di) {
      v8f r = acc[ci * 2 + di];
#pragma unroll
      for (int v = 0; v < 8; ++v) {
        int c = c0 + 16 * ci + v + 8 * g;
        int d = d0 + 16 * di + (lane & 15);
        lb[(size_t)c * C + d] = r[v] * 0.0625f;          // scale = C^-0.5
      }
    }
}

// ---------------------------------------------------------------------------
// Stage 3: row softmax (256 wide), f32 -> f16.  One wave32 per row.
// ---------------------------------------------------------------------------
__global__ void k_softmax(const float* __restrict__ lg, _Float16* __restrict__ at) {
  int row  = blockIdx.x * 8 + (threadIdx.x >> 5);
  int lane = threadIdx.x & 31;
  const float* rp = lg + (size_t)row * C;
  float v[8], m = -3.0e38f;
#pragma unroll
  for (int j = 0; j < 8; ++j) { v[j] = rp[lane + 32 * j]; m = fmaxf(m, v[j]); }
#pragma unroll
  for (int off = 16; off > 0; off >>= 1) m = fmaxf(m, __shfl_xor(m, off, 32));
  float s = 0.f;
#pragma unroll
  for (int j = 0; j < 8; ++j) { v[j] = __expf(v[j] - m); s += v[j]; }
#pragma unroll
  for (int off = 16; off > 0; off >>= 1) s += __shfl_xor(s, off, 32);
  float inv = 1.f / s;
  _Float16* op = at + (size_t)row * C;
#pragma unroll
  for (int j = 0; j < 8; ++j) op[lane + 32 * j] = (_Float16)(v[j] * inv);
}

// ---------------------------------------------------------------------------
// Stage 4: out2T[n,c] = sum_d attn[c,d]*VT[n,d]   (K = 256)
// ---------------------------------------------------------------------------
__global__ void k_out2(const _Float16* __restrict__ at, const _Float16* __restrict__ VT,
                       _Float16* __restrict__ o2) {
  int b    = blockIdx.z;
  int c0   = blockIdx.y * 16;
  int wave = threadIdx.x >> 5, lane = threadIdx.x & 31;
  int n0   = (blockIdx.x * 32 + wave * 4) * 16;
  const _Float16* ab = at + (size_t)b * C * C;
  const _Float16* vb = VT + (size_t)b * N * C;

  v8f acc[4] = {{}, {}, {}, {}};
  for (int k0 = 0; k0 < C; k0 += 32) {
    v16h a = load_a_frag(ab, C, c0, k0, lane);
#pragma unroll
    for (int t = 0; t < 4; ++t) {
      v16h bf = load_b_frag(vb, C, n0 + 16 * t, k0, lane);
      acc[t] = wmma_f(a, bf, acc[t]);
    }
  }
  _Float16* ob = o2 + (size_t)b * N * C;
  int g = lane >> 4;
#pragma unroll
  for (int t = 0; t < 4; ++t) {
    int n = n0 + 16 * t + (lane & 15);
    *(v8h*)(ob + (size_t)n * C + c0 + 8 * g) = cvt8(acc[t]);
  }
}

// ---------------------------------------------------------------------------
// Stage 5: proj.  P[m,n] = sum_c proj_w[m,c]*out2T[n,c] + proj_b[m]
// Written transposed into d_out (matches [B,N,C]->[B,C,H,W] flat reshape).
// ---------------------------------------------------------------------------
__global__ void k_proj(const _Float16* __restrict__ wp, const _Float16* __restrict__ o2,
                       const float* __restrict__ pb, float* __restrict__ out) {
  int b    = blockIdx.z;
  int m0   = blockIdx.y * 16;
  int wave = threadIdx.x >> 5, lane = threadIdx.x & 31;
  int n0   = (blockIdx.x * 32 + wave * 4) * 16;
  const _Float16* ob2 = o2 + (size_t)b * N * C;

  v8f acc[4] = {{}, {}, {}, {}};
  for (int k0 = 0; k0 < C; k0 += 32) {
    v16h a = load_a_frag(wp, C, m0, k0, lane);
#pragma unroll
    for (int t = 0; t < 4; ++t) {
      v16h bf = load_b_frag(ob2, C, n0 + 16 * t, k0, lane);
      acc[t] = wmma_f(a, bf, acc[t]);
    }
  }
  float* outb = out + (size_t)b * C * N;
  int g = lane >> 4;
#pragma unroll
  for (int t = 0; t < 4; ++t) {
    int n = n0 + 16 * t + (lane & 15);
    v8f r = acc[t];
#pragma unroll
    for (int v = 0; v < 8; ++v) r[v] += pb[m0 + 8 * g + v];
    *(v8f*)(outb + (size_t)n * C + m0 + 8 * g) = r;
  }
}

// ---------------------------------------------------------------------------
// Stage 6: 3x3 conv.  One block = one image row h x all 256 output channels.
// Per 32-wide K-chunk: stage a 3x66 halo'd token strip (12.7KB) into LDS once,
// then all 9 taps / 16 o-tiles / 4 p-tiles consume it (9x global-read reuse).
// Halo columns & out-of-image rows are zero in LDS -> no boundary branches
// anywhere near the WMMAs.  Each wave: 2 o-tiles x 4 p-tiles = 8 accumulators.
// ---------------------------------------------------------------------------
__global__ void k_conv(const _Float16* __restrict__ xt, const _Float16* __restrict__ w9,
                       const float* __restrict__ cb, float* __restrict__ out) {
  __shared__ _Float16 strip[3 * 66 * 32];                // [row 0..2][col 0..65][k 0..31]
  int h    = blockIdx.x;                                 // image row
  int b    = blockIdx.y;
  int wave = threadIdx.x >> 5, lane = threadIdx.x & 31;
  const _Float16* xb = xt + (size_t)b * N * C;

  // Zero the halo columns (c==0, c==65) once; they are never overwritten.
  if (threadIdx.x < 192) {
    int r = threadIdx.x / 64, rem = threadIdx.x % 64;
    int c = (rem < 32) ? 0 : 65;
    strip[(r * 66 + c) * 32 + (rem & 31)] = (_Float16)0.f;
  }

  v8f acc[8] = {{}, {}, {}, {}, {}, {}, {}, {}};         // [t][oi]
  for (int k0 = 0; k0 < C; k0 += 32) {
    // Stage interior: 3 rows x 64 cols x 4 chunks of 8 halves = 768 units.
#pragma unroll
    for (int s = 0; s < 3; ++s) {
      int u = threadIdx.x + 256 * s;
      int j = u & 3, c = (u >> 2) & 63, r = u >> 8;
      int hs = h + r - 1;
      v8h val;
      if ((unsigned)hs < 64u) {
        val = *(const v8h*)(xb + (size_t)(hs * 64 + c) * C + k0 + j * 8);
      } else {
#pragma unroll
        for (int i = 0; i < 8; ++i) val[i] = (_Float16)0.f;
      }
      *(v8h*)(&strip[(r * 66 + (c + 1)) * 32 + j * 8]) = val;
    }
    __syncthreads();

    for (int kk = 0; kk < 9; ++kk) {
      int dh = kk / 3 - 1, dw = kk % 3 - 1;
      const _Float16* wk = w9 + (size_t)kk * C * C;
      v16h a0 = load_a_frag(wk, C, wave * 32,      k0, lane);
      v16h a1 = load_a_frag(wk, C, wave * 32 + 16, k0, lane);
#pragma unroll
      for (int t = 0; t < 4; ++t) {
        int ccol = 16 * t + (lane & 15) + dw + 1;        // strip column (halo offset +1)
        const _Float16* p = &strip[((dh + 1) * 66 + ccol) * 32 + (lane >> 4) * 16];
        v16h bf = combine(*(const v8h*)p, *(const v8h*)(p + 8));
        acc[t * 2]     = wmma_f(a0, bf, acc[t * 2]);
        acc[t * 2 + 1] = wmma_f(a1, bf, acc[t * 2 + 1]);
      }
    }
    __syncthreads();
  }

  float* outb = out + (size_t)b * C * N;
  int g = lane >> 4;
#pragma unroll
  for (int t = 0; t < 4; ++t)
#pragma unroll
    for (int oi = 0; oi < 2; ++oi) {
      v8f r = acc[t * 2 + oi];
      int p = h * 64 + 16 * t + (lane & 15);
#pragma unroll
      for (int v = 0; v < 8; ++v) {
        int o = wave * 32 + oi * 16 + v + 8 * g;
        outb[(size_t)o * N + p] += r[v] + cb[o];
      }
    }
}

// ---------------------------------------------------------------------------
// Launch
// ---------------------------------------------------------------------------
extern "C" void kernel_launch(void* const* d_in, const int* in_sizes, int n_in,
                              void* d_out, int out_size, void* d_ws, size_t ws_size,
                              hipStream_t stream) {
  (void)in_sizes; (void)n_in; (void)out_size; (void)ws_size;
  const float* x      = (const float*)d_in[0];
  const float* qkv_w  = (const float*)d_in[1];
  const float* qkv_b  = (const float*)d_in[2];
  const float* proj_w = (const float*)d_in[3];
  const float* proj_b = (const float*)d_in[4];
  const float* conv_w = (const float*)d_in[5];
  const float* conv_b = (const float*)d_in[6];
  float* out = (float*)d_out;

  char* w = (char*)d_ws;
  size_t off = 0;
  auto alloc = [&](size_t bytes) -> void* {
    void* p = w + off;
    off = (off + bytes + 255) & ~(size_t)255;
    return p;
  };
  _Float16* XT = (_Float16*)alloc((size_t)B * N * C * 2);   // x transposed, f16
  _Float16* WQ = (_Float16*)alloc((size_t)3 * C * C * 2);   // qkv_w f16
  _Float16* WP = (_Float16*)alloc((size_t)C * C * 2);       // proj_w f16
  _Float16* W9 = (_Float16*)alloc((size_t)9 * C * C * 2);   // conv_w [kk][O][I] f16
  _Float16* Qm = (_Float16*)alloc((size_t)B * C * N * 2);   // Q [c][n]
  _Float16* KT = (_Float16*)alloc((size_t)B * C * N * 2);   // K^T [d][n]
  _Float16* VT = (_Float16*)alloc((size_t)B * N * C * 2);   // V^T [n][d]
  float*    LG = (float*)   alloc((size_t)B * C * C * 4);   // attn logits f32
  _Float16* AT = (_Float16*)alloc((size_t)B * C * C * 2);   // attn f16
  _Float16* O2 = (_Float16*)alloc((size_t)B * N * C * 2);   // out2T [n][c]

  k_pack_xt   <<<dim3(N / 16, C / 16, B), dim3(16, 16), 0, stream>>>(x, XT);
  k_pack_w    <<<(3 * C * C + 255) / 256, 256, 0, stream>>>(qkv_w, WQ, 3 * C * C);
  k_pack_w    <<<(C * C + 255) / 256, 256, 0, stream>>>(proj_w, WP, C * C);
  k_pack_convw<<<(9 * C * C + 255) / 256, 256, 0, stream>>>(conv_w, W9);

  k_qkv    <<<dim3(8, 48, B), 256, 0, stream>>>(XT, WQ, qkv_b, Qm, KT, VT);
  k_attn   <<<dim3(8, B),     256, 0, stream>>>(Qm, KT, LG);
  k_softmax<<<(B * C) / 8,    256, 0, stream>>>(LG, AT);
  k_out2   <<<dim3(8, 16, B), 256, 0, stream>>>(AT, VT, O2);
  k_proj   <<<dim3(8, 16, B), 256, 0, stream>>>(WP, O2, proj_b, out);
  k_conv   <<<dim3(64, B),    256, 0, stream>>>(XT, W9, conv_b, out);
}
